// Mamba_46772193853553
// MI455X (gfx1250) — matrix operand order repaired and checked by reading, not compiled, not run on hardware
//
#include <hip/hip_runtime.h>
#include <hip/hip_bf16.h>

typedef __attribute__((ext_vector_type(8)))  __bf16 v8bf;
typedef __attribute__((ext_vector_type(16))) __bf16 v16bf;
typedef __attribute__((ext_vector_type(8)))  float  v8f;
typedef int v4i_ __attribute__((vector_size(16)));

#define B_    4
#define L_    4096
#define DM    1024
#define DI    2048
#define DS    64
#define BLROWS (B_ * L_)          // 16384
#define CHUNK  128
#define NCH    (L_ / CHUNK)       // 32
#define LDT    40                 // padded LDS row stride in bf16 (80B, 16B-aligned)

// ---------------------------------------------------------------------------
// CDNA5 async global->LDS copy (ASYNCcnt path), with sync fallback
// ---------------------------------------------------------------------------
#if __has_builtin(__builtin_amdgcn_global_load_async_to_lds_b128)
#  define MAMBA_ASYNC_LDS 1
#else
#  define MAMBA_ASYNC_LDS 0
#endif

#if MAMBA_ASYNC_LDS
__device__ __forceinline__ void asy_b128(void* lds, const void* g)
{
    // (v4i __device__* gaddr, v4i __shared__* lds, Iint off, Iint cpol)
    __builtin_amdgcn_global_load_async_to_lds_b128((v4i_*)g, (v4i_*)lds, 0, 0);
}
#endif

__device__ __forceinline__ void wait_async0()
{
#if __has_builtin(__builtin_amdgcn_s_wait_asynccnt)
    __builtin_amdgcn_s_wait_asynccnt(0);
#else
    asm volatile("s_wait_asynccnt 0x0" ::: "memory");
#endif
}

#if __has_builtin(__builtin_amdgcn_sched_barrier)
#  define SCHED_FENCE() __builtin_amdgcn_sched_barrier(0)
#else
#  define SCHED_FENCE() ((void)0)
#endif

// stage ROWS x 32 bf16 tile into LDS (16B chunk per thread per iter)
template<int ROWS>
__device__ __forceinline__ void stage_tile(__bf16* lds, const __bf16* __restrict__ g,
                                           int KD, int rbase, int k0, int tid)
{
    constexpr int IT = (ROWS * 4) / 256;
#pragma unroll
    for (int it = 0; it < IT; ++it) {
        const int idx = tid + it * 256;
        const int r = idx >> 2, c = (idx & 3) * 8;
#if MAMBA_ASYNC_LDS
        asy_b128(&lds[r * LDT + c], &g[(long)(rbase + r) * KD + k0 + c]);
#else
        *(v8bf*)&lds[r * LDT + c] = *(const v8bf*)&g[(long)(rbase + r) * KD + k0 + c];
#endif
    }
}

// ---------------------------------------------------------------------------
// bf16 WMMA GEMM core: C[BM x BN] += A[BM x KD] * W[BN x KD]^T
// 256 threads = 8 waves; wave tile = 32(M) x 64(N): 2 A-frags x 4 B-frags
//   -> acc[2][4] of v8f, 8 v_wmma per K-step.
// Double-buffered LDS with async copies overlapping WMMA; KD/32 must be even.
// ---------------------------------------------------------------------------
template<int WAVES_M, int WAVES_N>
__device__ __forceinline__ void gemm_acc_bf16(const __bf16* __restrict__ A,
                                              const __bf16* __restrict__ W,
                                              int KD, int mBase, int nBase,
                                              v8f (&acc)[2][4])
{
    constexpr int BM = WAVES_M * 32;
    constexpr int BN = WAVES_N * 64;
    __shared__ __attribute__((aligned(16))) __bf16 lA[2][BM * LDT];
    __shared__ __attribute__((aligned(16))) __bf16 lW[2][BN * LDT];

    const int tid  = threadIdx.x;
    const int wave = tid >> 5;
    const int lane = tid & 31;
    const int half = lane >> 4;
    const int lr   = lane & 15;
    const int wm   = wave % WAVES_M;
    const int wn   = wave / WAVES_M;

    v8f z = {};
#pragma unroll
    for (int a = 0; a < 2; ++a)
#pragma unroll
        for (int f = 0; f < 4; ++f) acc[a][f] = z;

    auto stage = [&](int buf, int k0) {
        stage_tile<BM>(lA[buf], A, KD, mBase, k0, tid);
        stage_tile<BN>(lW[buf], W, KD, nBase, k0, tid);
    };

    auto compute = [&](const __bf16* bA, const __bf16* bW) {
        v16bf af[2];
#pragma unroll
        for (int a = 0; a < 2; ++a) {
            const int ar = (wm * 32 + a * 16 + lr) * LDT;
            v8bf alo = *(const v8bf*)&bA[ar + half * 8];
            v8bf ahi = *(const v8bf*)&bA[ar + 16 + half * 8];
            af[a] = __builtin_shufflevector(alo, ahi,
                0,1,2,3,4,5,6,7,8,9,10,11,12,13,14,15);
        }
#pragma unroll
        for (int f = 0; f < 4; ++f) {
            const int br = (wn * 64 + f * 16 + lr) * LDT;
            v8bf blo = *(const v8bf*)&bW[br + half * 8];
            v8bf bhi = *(const v8bf*)&bW[br + 16 + half * 8];
            v16bf bfrag = __builtin_shufflevector(blo, bhi,
                0,1,2,3,4,5,6,7,8,9,10,11,12,13,14,15);
            acc[0][f] = __builtin_amdgcn_wmma_f32_16x16x32_bf16(
                false, af[0], false, bfrag, (short)0, acc[0][f], false, false);
            acc[1][f] = __builtin_amdgcn_wmma_f32_16x16x32_bf16(
                false, af[1], false, bfrag, (short)0, acc[1][f], false, false);
            SCHED_FENCE();   // keep fragment live ranges short; acc stays home
        }
    };

    const int nsteps = KD / 32;   // even for all GEMMs in this model
#if MAMBA_ASYNC_LDS
    stage(0, 0);
    for (int s = 0; s < nsteps; s += 2) {
        wait_async0();
        __syncthreads();
        stage(1, (s + 1) * 32);            // s+1 < nsteps (nsteps even)
        compute(lA[0], lW[0]);
        wait_async0();
        __syncthreads();
        if (s + 2 < nsteps) stage(0, (s + 2) * 32);
        compute(lA[1], lW[1]);
    }
#else
    for (int s = 0; s < nsteps; ++s) {
        stage(0, s * 32);
        __syncthreads();
        compute(lA[0], lW[0]);
        __syncthreads();
    }
#endif
}

// ---------------------------------------------------------------------------
// f32 -> bf16 conversion (one element per thread)
// ---------------------------------------------------------------------------
__global__ void mamba_cvt_bf16(const float* __restrict__ s, __bf16* __restrict__ d, long n)
{
    long i = (long)blockIdx.x * blockDim.x + threadIdx.x;
    if (i < n) d[i] = (__bf16)s[i];
}

// ---------------------------------------------------------------------------
// in_proj: [16384,1024] x [4096,1024]^T -> u(bf16), gate(bf16) each [16384,2048]
// ---------------------------------------------------------------------------
__global__ __launch_bounds__(256, 1)
void mamba_gemm_inproj(const __bf16* __restrict__ xb, const __bf16* __restrict__ win,
                       const float* __restrict__ bias,
                       __bf16* __restrict__ u, __bf16* __restrict__ gate)
{
    const int mBase = blockIdx.x * 128, nBase = blockIdx.y * 128;
    v8f acc[2][4];
    gemm_acc_bf16<4, 2>(xb, win, DM, mBase, nBase, acc);

    const int tid = threadIdx.x, wave = tid >> 5, lane = tid & 31;
    const int half = lane >> 4, lr = lane & 15;
    const int wm = wave % 4, wn = wave / 4;
#pragma unroll
    for (int a = 0; a < 2; ++a)
#pragma unroll
    for (int f = 0; f < 4; ++f) {
        const int n = nBase + wn * 64 + f * 16 + lr;
        const float bn = bias[n];
#pragma unroll
        for (int r = 0; r < 8; ++r) {
            const int m = mBase + wm * 32 + a * 16 + half * 8 + r;
            float v = acc[a][f][r] + bn;
            if (n < DI) u[(long)m * DI + n] = (__bf16)v;
            else        gate[(long)m * DI + (n - DI)] = (__bf16)v;
        }
    }
}

// ---------------------------------------------------------------------------
// causal depthwise conv1d (k=4) + SiLU : u(bf16) -> uc(bf16)
// ---------------------------------------------------------------------------
__global__ __launch_bounds__(256)
void mamba_conv_silu(const __bf16* __restrict__ u, const float* __restrict__ cw,
                     const float* __restrict__ cb, __bf16* __restrict__ uc)
{
    const long gid = (long)blockIdx.x * 256 + threadIdx.x;
    const int  CH8 = DI / 8;
    const long bl  = gid / CH8;
    if (bl >= (long)BLROWS) return;
    const int c8 = (int)(gid % CH8) * 8;
    const int l  = (int)(bl % L_);
    const long b = bl / L_;

    float accv[8];
#pragma unroll
    for (int i = 0; i < 8; ++i) accv[i] = cb[c8 + i];

#pragma unroll
    for (int j = 0; j < 4; ++j) {
        const int ls = l - 3 + j;
        if (ls < 0) continue;
        v8bf uv = *(const v8bf*)&u[((b * L_ + ls) * DI) + c8];
#pragma unroll
        for (int i = 0; i < 8; ++i)
            accv[i] += (float)uv[i] * cw[(c8 + i) * 4 + j];
    }
    v8bf o;
#pragma unroll
    for (int i = 0; i < 8; ++i) {
        float s = accv[i];
        o[i] = (__bf16)(s / (1.f + expf(-s)));
    }
    *(v8bf*)&uc[(b * L_ + l) * DI + c8] = o;
}

// ---------------------------------------------------------------------------
// dt/B/C projections in one GEMM: [16384,2048] x [192,2048]^T
// rows 0-63 of W = dt_w, 64-127 = B_w, 128-191 = C_w
// ---------------------------------------------------------------------------
__global__ __launch_bounds__(256, 1)
void mamba_gemm_dbc(const __bf16* __restrict__ ucb, const __bf16* __restrict__ wdbc,
                    const float* __restrict__ dt_b,
                    float* __restrict__ dt, float* __restrict__ Bt, float* __restrict__ Ct)
{
    const int mBase = blockIdx.x * 256, nBase = blockIdx.y * 64;
    v8f acc[2][4];
    gemm_acc_bf16<8, 1>(ucb, wdbc, DI, mBase, nBase, acc);

    const int tid = threadIdx.x, wave = tid >> 5, lane = tid & 31;
    const int half = lane >> 4, lr = lane & 15;
    const int wm = wave;                            // WAVES_N == 1
#pragma unroll
    for (int a = 0; a < 2; ++a)
#pragma unroll
    for (int f = 0; f < 4; ++f) {
        const int n = nBase + f * 16 + lr;          // 0..191
        const int part = n >> 6, col = n & 63;
#pragma unroll
        for (int r = 0; r < 8; ++r) {
            const int m = mBase + wm * 32 + a * 16 + half * 8 + r;
            float v = acc[a][f][r];
            if (part == 0) {
                v += dt_b[col];
                float sp = (v > 20.f) ? v : log1pf(expf(v));   // softplus
                sp = fminf(fmaxf(sp, 1e-3f), 1e-1f);           // clip
                dt[(long)m * DS + col] = sp;
            } else if (part == 1) {
                Bt[(long)m * DS + col] = v;
            } else {
                Ct[(long)m * DS + col] = v;
            }
        }
    }
}

// ---------------------------------------------------------------------------
// selective scan, 3-phase chunked
// ---------------------------------------------------------------------------
__global__ void mamba_scan_chunk(const float* __restrict__ dt, const float* __restrict__ Bt,
                                 const float* __restrict__ A_log,
                                 float* __restrict__ hloc, float* __restrict__ pexp,
                                 float* __restrict__ decay, float* __restrict__ hend)
{
    const int b = blockIdx.x, ch = blockIdx.y, s = threadIdx.x;
    const float aS = -expf(A_log[s]);
    float lp = 0.f, h = 0.f;
    const long base = ((long)b * L_ + (long)ch * CHUNK) * DS + s;
    for (int i = 0; i < CHUNK; ++i) {
        const long idx = base + (long)i * DS;
        const float d  = dt[idx];
        const float la = aS * d;
        h  = h * expf(la) + Bt[idx] * d;
        lp += la;
        hloc[idx] = h;
        pexp[idx] = expf(lp);
    }
    const long ci = ((long)b * NCH + ch) * DS + s;
    decay[ci] = expf(lp);
    hend[ci]  = h;
}

__global__ void mamba_scan_combine(const float* __restrict__ decay,
                                   const float* __restrict__ hend,
                                   float* __restrict__ carry)
{
    const int t = threadIdx.x;                 // 256 = 4 batches * 64 states
    const int b = t >> 6, s = t & 63;
    float H = 0.f;
    for (int ch = 0; ch < NCH; ++ch) {
        const long ci = ((long)b * NCH + ch) * DS + s;
        carry[ci] = H;
        H = hend[ci] + decay[ci] * H;
    }
}

__global__ void mamba_scan_fix(const float* __restrict__ hloc, const float* __restrict__ pexp,
                               const float* __restrict__ carry, const float* __restrict__ Ct,
                               __bf16* __restrict__ ys)
{
    const long i = (long)blockIdx.x * 256 + threadIdx.x;
    if (i >= (long)BLROWS * DS) return;
    const int  s  = (int)(i & 63);
    const long bl = i >> 6;
    const int  l  = (int)(bl % L_);
    const int  b  = (int)(bl / L_);
    const int  ch = l / CHUNK;
    const float h = hloc[i] + pexp[i] * carry[((long)b * NCH + ch) * DS + s];
    ys[i] = (__bf16)(h * Ct[i]);
}

// ---------------------------------------------------------------------------
// os GEMM + skip + gate: yin = (ys @ os_w^T + uc*D) * sigmoid(gate)  (bf16 out)
// ---------------------------------------------------------------------------
__global__ __launch_bounds__(256, 1)
void mamba_gemm_os(const __bf16* __restrict__ ys, const __bf16* __restrict__ wos,
                   const __bf16* __restrict__ ucb, const __bf16* __restrict__ gate,
                   const float* __restrict__ Dp, __bf16* __restrict__ yin)
{
    const int mBase = blockIdx.x * 128, nBase = blockIdx.y * 128;
    v8f acc[2][4];
    gemm_acc_bf16<4, 2>(ys, wos, DS, mBase, nBase, acc);

    const int tid = threadIdx.x, wave = tid >> 5, lane = tid & 31;
    const int half = lane >> 4, lr = lane & 15;
    const int wm = wave % 4, wn = wave / 4;
#pragma unroll
    for (int a = 0; a < 2; ++a)
#pragma unroll
    for (int f = 0; f < 4; ++f) {
        const int n = nBase + wn * 64 + f * 16 + lr;
        const float dn = Dp[n];
#pragma unroll
        for (int r = 0; r < 8; ++r) {
            const int m = mBase + wm * 32 + a * 16 + half * 8 + r;
            const long o = (long)m * DI + n;
            float y = acc[a][f][r] + (float)ucb[o] * dn;
            const float g = (float)gate[o];
            y *= 1.f / (1.f + expf(-g));
            yin[o] = (__bf16)y;
        }
    }
}

// ---------------------------------------------------------------------------
// out GEMM: [16384,2048] x [1024,2048]^T + out_b -> f32 d_out
// ---------------------------------------------------------------------------
__global__ __launch_bounds__(256, 1)
void mamba_gemm_out(const __bf16* __restrict__ yin, const __bf16* __restrict__ wout,
                    const float* __restrict__ ob, float* __restrict__ out)
{
    const int mBase = blockIdx.x * 128, nBase = blockIdx.y * 128;
    v8f acc[2][4];
    gemm_acc_bf16<4, 2>(yin, wout, DI, mBase, nBase, acc);

    const int tid = threadIdx.x, wave = tid >> 5, lane = tid & 31;
    const int half = lane >> 4, lr = lane & 15;
    const int wm = wave % 4, wn = wave / 4;
#pragma unroll
    for (int a = 0; a < 2; ++a)
#pragma unroll
    for (int f = 0; f < 4; ++f) {
        const int n = nBase + wn * 64 + f * 16 + lr;
        const float bn = ob[n];
#pragma unroll
        for (int r = 0; r < 8; ++r) {
            const int m = mBase + wm * 32 + a * 16 + half * 8 + r;
            out[(long)m * DM + n] = acc[a][f][r] + bn;
        }
    }
}

// ---------------------------------------------------------------------------
extern "C" void kernel_launch(void* const* d_in, const int* in_sizes, int n_in,
                              void* d_out, int out_size, void* d_ws, size_t ws_size,
                              hipStream_t stream)
{
    (void)in_sizes; (void)n_in; (void)out_size; (void)ws_size;
    const float* x    = (const float*)d_in[0];
    const float* w_in = (const float*)d_in[1];
    const float* b_in = (const float*)d_in[2];
    const float* c_w  = (const float*)d_in[3];
    const float* c_b  = (const float*)d_in[4];
    const float* B_w  = (const float*)d_in[5];
    const float* C_w  = (const float*)d_in[6];
    const float* dt_w = (const float*)d_in[7];
    const float* dt_b = (const float*)d_in[8];
    const float* A_lg = (const float*)d_in[9];
    const float* D_p  = (const float*)d_in[10];
    const float* os_w = (const float*)d_in[11];
    const float* o_w  = (const float*)d_in[12];
    const float* o_b  = (const float*)d_in[13];
    float* out = (float*)d_out;

    // workspace carve-up (256B aligned)
    char* ws = (char*)d_ws;
    size_t off = 0;
    auto carve = [&](size_t bytes) {
        char* p = ws + off;
        off = (off + bytes + 255) & ~(size_t)255;
        return p;
    };
    __bf16* xb    = (__bf16*)carve((size_t)BLROWS * DM * 2);
    __bf16* win   = (__bf16*)carve((size_t)2 * DI * DM * 2);
    __bf16* wdbc  = (__bf16*)carve((size_t)192 * DI * 2);
    __bf16* wos   = (__bf16*)carve((size_t)DI * DS * 2);
    __bf16* wout  = (__bf16*)carve((size_t)DM * DI * 2);
    __bf16* u_bf  = (__bf16*)carve((size_t)BLROWS * DI * 2);
    __bf16* g_bf  = (__bf16*)carve((size_t)BLROWS * DI * 2);
    __bf16* uc_bf = (__bf16*)carve((size_t)BLROWS * DI * 2);
    float*  dt_f  = (float*) carve((size_t)BLROWS * DS * 4);
    float*  Bt_f  = (float*) carve((size_t)BLROWS * DS * 4);
    float*  Ct_f  = (float*) carve((size_t)BLROWS * DS * 4);
    float*  hloc  = (float*) carve((size_t)BLROWS * DS * 4);
    float*  pexp  = (float*) carve((size_t)BLROWS * DS * 4);
    float*  decay = (float*) carve((size_t)B_ * NCH * DS * 4);
    float*  hend  = (float*) carve((size_t)B_ * NCH * DS * 4);
    float*  carry = (float*) carve((size_t)B_ * NCH * DS * 4);
    __bf16* ys_bf = (__bf16*)carve((size_t)BLROWS * DS * 2);
    __bf16* yin   = (__bf16*)carve((size_t)BLROWS * DI * 2);

    auto cvt = [&](const float* s, __bf16* d, long n) {
        mamba_cvt_bf16<<<(unsigned)((n + 255) / 256), 256, 0, stream>>>(s, d, n);
    };
    cvt(x,    xb,   (long)BLROWS * DM);
    cvt(w_in, win,  (long)2 * DI * DM);
    cvt(dt_w, wdbc + 0 * (size_t)DS * DI, (long)DS * DI);
    cvt(B_w,  wdbc + 1 * (size_t)DS * DI, (long)DS * DI);
    cvt(C_w,  wdbc + 2 * (size_t)DS * DI, (long)DS * DI);
    cvt(os_w, wos,  (long)DI * DS);
    cvt(o_w,  wout, (long)DM * DI);

    // in_proj GEMM -> u, gate
    mamba_gemm_inproj<<<dim3(BLROWS / 128, (2 * DI) / 128), 256, 0, stream>>>(
        xb, win, b_in, u_bf, g_bf);

    // conv + silu
    mamba_conv_silu<<<(unsigned)(((long)BLROWS * (DI / 8) + 255) / 256), 256, 0, stream>>>(
        u_bf, c_w, c_b, uc_bf);

    // dt / B / C projections
    mamba_gemm_dbc<<<dim3(BLROWS / 256, 192 / 64), 256, 0, stream>>>(
        uc_bf, wdbc, dt_b, dt_f, Bt_f, Ct_f);

    // 3-phase selective scan
    mamba_scan_chunk<<<dim3(B_, NCH), DS, 0, stream>>>(dt_f, Bt_f, A_lg, hloc, pexp, decay, hend);
    mamba_scan_combine<<<1, B_ * DS, 0, stream>>>(decay, hend, carry);
    mamba_scan_fix<<<(unsigned)(((long)BLROWS * DS + 255) / 256), 256, 0, stream>>>(
        hloc, pexp, carry, Ct_f, ys_bf);

    // os GEMM + skip + gate
    mamba_gemm_os<<<dim3(BLROWS / 128, DI / 128), 256, 0, stream>>>(
        ys_bf, wos, uc_bf, g_bf, D_p, yin);

    // out GEMM -> f32 output
    mamba_gemm_out<<<dim3(BLROWS / 128, DM / 128), 256, 0, stream>>>(
        yin, wout, o_b, out);
}
